// SimpleLSTM_19971597926679
// MI455X (gfx1250) — compile-verified, matrix operand order checked by reading, not run on hardware
//
#include <hip/hip_runtime.h>
#include <hip/hip_bf16.h>

typedef __bf16 bf16;
typedef __attribute__((ext_vector_type(16))) __bf16 v16bf;
typedef __attribute__((ext_vector_type(8)))  __bf16 v8bf;
typedef __attribute__((ext_vector_type(8)))  float  v8f;
typedef __attribute__((__vector_size__(16))) int    v4i;

#define AS1 __attribute__((address_space(1)))
#define AS3 __attribute__((address_space(3)))

#define BSZ   32
#define SLEN  2048
#define DIM   256
#define HDIM  512
#define G4H   2048
#define ODIM  256
#define BS    (BSZ * SLEN)        // 65536
#define NWG   32                  // persistent grid size

#define WH_PAD        8                     // bf16 pad -> stride 520 elems (1040B)
#define WH_STRIDE     (HDIM + WH_PAD)       // conflict-free ds_load_b128 pattern
#define WH_LDS_BYTES  (64 * WH_STRIDE * sizeof(bf16))

union FragU { v16bf v; v8bf h[2]; };

// Load a 16x32 bf16 WMMA A/B fragment from row-major [rows, K] storage.
// ISA layout (16-bit A/B, wave32): lanes 0-15 hold K=[kb..kb+7] (v0-3) and
// K=[kb+16..kb+23] (v4-7); lanes 16-31 hold K=[kb+8..kb+15], K=[kb+24..kb+31].
__device__ inline v16bf load_frag(const bf16* __restrict__ base, int row, int ld,
                                  int kb, int lane) {
  const int c0 = (lane >> 4) << 3;                 // 0 or 8
  const bf16* p = base + (size_t)row * ld + kb + c0;
  FragU f;
  f.h[0] = *(const v8bf*)(p);
  f.h[1] = *(const v8bf*)(p + 16);
  return f.v;
}

// Same fragment gather but from an LDS-resident, padded [64][WH_STRIDE] tile.
__device__ inline v16bf load_frag_lds(const bf16* base, int row_off, int kb, int lane) {
  const int c0 = (lane >> 4) << 3;
  const bf16* p = base + row_off + kb + c0;
  FragU f;
  f.h[0] = *(const v8bf*)(p);
  f.h[1] = *(const v8bf*)(p + 16);
  return f.v;
}

// CDNA5 async global->LDS copy of one 16-byte chunk (tracked by ASYNCcnt).
__device__ inline void async_copy16(const bf16* src, bf16* dst) {
#if __has_builtin(__builtin_amdgcn_global_load_async_to_lds_b128)
  __builtin_amdgcn_global_load_async_to_lds_b128(
      (AS1 v4i*)src, (AS3 v4i*)dst, 0, 0);
#else
  *(v8bf*)dst = *(const v8bf*)src;
#endif
}

__device__ inline void async_copy_drain() {
#if __has_builtin(__builtin_amdgcn_global_load_async_to_lds_b128)
#if __has_builtin(__builtin_amdgcn_s_wait_asynccnt)
  __builtin_amdgcn_s_wait_asynccnt(0);
#else
  asm volatile("s_wait_asynccnt 0x0" ::: "memory");
#endif
#endif
}

__device__ inline float sigmoidf_fast(float x) {
  return 1.0f / (1.0f + __expf(-x));
}

// ---------------- prep kernels ----------------

__global__ void init_kernel(bf16* __restrict__ hbuf, unsigned* __restrict__ ctr) {
  size_t i = (size_t)blockIdx.x * blockDim.x + threadIdx.x;
  if (i == 0) *ctr = 0u;
  if (i < (size_t)2 * BSZ * HDIM) hbuf[i] = (bf16)0.0f;
}

__global__ void cast_bf16_kernel(const float* __restrict__ in, bf16* __restrict__ out,
                                 size_t n) {
  size_t i = (size_t)blockIdx.x * blockDim.x + threadIdx.x;
  if (i < n) out[i] = (bf16)in[i];
}

// out[n*K + k] = (bf16) in[k*N + n]   (transpose + downconvert)
__global__ void transpose_cast_kernel(const float* __restrict__ in, bf16* __restrict__ out,
                                      int K, int N) {
  int n = blockIdx.x * 16 + (threadIdx.x & 15);
  int k = blockIdx.y * 16 + (threadIdx.x >> 4);
  if (n < N && k < K) out[(size_t)n * K + k] = (bf16)in[(size_t)k * N + n];
}

// ---------------- phase 1: xg = x @ Wx  (f32 out) ----------------
// grid: (BS/16, 4H/(16*8*2)), block 256; each wave computes a 16x32 strip
// (two N-tiles sharing one A fragment -> half the A traffic, 2x WMMA ILP).
__global__ __launch_bounds__(256) void gemm_xg_kernel(const bf16* __restrict__ xb,
                                                      const bf16* __restrict__ wxT,
                                                      float* __restrict__ xg) {
  const int lane = threadIdx.x & 31, wave = threadIdx.x >> 5;
  const int m0 = blockIdx.x * 16;
  const int n0 = (blockIdx.y * 8 + wave) * 32;
  const int arow = m0 + (lane & 15);
  const int brow = n0 + (lane & 15);
  v8f acc0 = {}, acc1 = {};
  for (int kb = 0; kb < DIM; kb += 32) {
    v16bf a  = load_frag(xb,  arow,      DIM, kb, lane);
    v16bf b0 = load_frag(wxT, brow,      DIM, kb, lane);
    v16bf b1 = load_frag(wxT, brow + 16, DIM, kb, lane);
    acc0 = __builtin_amdgcn_wmma_f32_16x16x32_bf16(false, a, false, b0, (short)0, acc0,
                                                   false, false);
    acc1 = __builtin_amdgcn_wmma_f32_16x16x32_bf16(false, a, false, b1, (short)0, acc1,
                                                   false, false);
  }
  float* out = xg + (size_t)(m0 + ((lane >> 4) << 3)) * G4H + n0 + (lane & 15);
#pragma unroll
  for (int v = 0; v < 8; ++v) {
    out[(size_t)v * G4H]      = acc0[v];
    out[(size_t)v * G4H + 16] = acc1[v];
  }
}

// ---------------- phase 2: persistent LSTM recurrence ----------------
// grid: exactly NWG blocks of 256 threads. Block w owns h-columns [16w,16w+16).
// Its Wh slice (64 gate-columns x 512 K, bf16) is staged once into LDS via the
// CDNA5 async global->LDS path, then every timestep's B operand is ds_load'ed.
__global__ __launch_bounds__(256) void lstm_kernel(const float* __restrict__ xg,
                                                   const bf16* __restrict__ whT,
                                                   const float* __restrict__ bias,
                                                   bf16* __restrict__ hbuf,
                                                   bf16* __restrict__ hs,
                                                   unsigned* __restrict__ ctr) {
  extern __shared__ __align__(16) bf16 wh_lds[];   // [64][WH_STRIDE]
  __shared__ float gate_s[4][BSZ][16];   // i,f,g,o tiles exchanged between waves
  __shared__ float c_s[BSZ][16];         // cell state, resident in LDS all 2048 steps

  const int tid  = threadIdx.x;
  const int lane = tid & 31, wave = tid >> 5;
  const int wg   = blockIdx.x;
  const int n0   = wg * 16;              // h-column strip
  const int mt   = wave & 1;             // M tile (batch rows 0-15 / 16-31)
  const int gate = wave >> 1;            // 0=i 1=f 2=g 3=o

  for (int e = tid; e < BSZ * 16; e += 256) c_s[e >> 4][e & 15] = 0.0f;

  // ---- stage Wh slice into LDS: local col lc = g*16+j <-> WhT row g*512+n0+j
  for (int idx = tid; idx < 64 * (HDIM / 8); idx += 256) {   // 4096 x 16B chunks
    const int lc = idx >> 6;            // 64 chunks per column
    const int ck = (idx & 63) * 8;      // k element offset
    const int g = lc >> 4, j = lc & 15;
    const bf16* src = whT + (size_t)(g * HDIM + n0 + j) * HDIM + ck;
    bf16* dst = wh_lds + lc * WH_STRIDE + ck;
    async_copy16(src, dst);
  }
  async_copy_drain();
  __syncthreads();

  const int arow   = mt * 16 + (lane & 15);                    // batch row, A frag
  const int brow_o = (gate * 16 + (lane & 15)) * WH_STRIDE;    // LDS row offset, B frag

  for (int t = 0; t < SLEN; ++t) {
    const bf16* hcur = hbuf + (size_t)(t & 1) * (BSZ * HDIM);
    bf16* hnxt       = hbuf + (size_t)((t + 1) & 1) * (BSZ * HDIM);

    // prefetch next step's xg slice (streams from HBM; hide latency)
    if (t + 1 < SLEN && tid < 128) {
      int g = tid >> 5, r = tid & 31;
      __builtin_prefetch(&xg[((size_t)r * SLEN + (t + 1)) * G4H + g * HDIM + n0], 0, 0);
    }

    // gates(tile) = h(t) @ Wh   — K=512: A from global (L2-hot), B from LDS
    v8f acc = {};
    for (int kb = 0; kb < HDIM; kb += 32) {
      v16bf a = load_frag(hcur, arow, HDIM, kb, lane);
      v16bf b = load_frag_lds(wh_lds, brow_o, kb, lane);
      acc = __builtin_amdgcn_wmma_f32_16x16x32_bf16(false, a, false, b, (short)0, acc,
                                                    false, false);
    }
    {
      const int r0 = mt * 16 + ((lane >> 4) << 3);
      const int cc = lane & 15;
#pragma unroll
      for (int v = 0; v < 8; ++v) gate_s[gate][r0 + v][cc] = acc[v];
    }
    __syncthreads();

    // fused elementwise LSTM cell (f32): 512 elements over 256 threads
    for (int e = tid; e < BSZ * 16; e += 256) {
      const int r = e >> 4, cc = e & 15;
      const size_t xrow = (size_t)r * SLEN + t;               // b*S + t
      const float gi = gate_s[0][r][cc] + xg[xrow * G4H +          n0 + cc] + bias[         n0 + cc];
      const float gf = gate_s[1][r][cc] + xg[xrow * G4H +  HDIM  + n0 + cc] + bias[ HDIM  + n0 + cc];
      const float gg = gate_s[2][r][cc] + xg[xrow * G4H + 2*HDIM + n0 + cc] + bias[2*HDIM + n0 + cc];
      const float go = gate_s[3][r][cc] + xg[xrow * G4H + 3*HDIM + n0 + cc] + bias[3*HDIM + n0 + cc];
      const float cn = sigmoidf_fast(gf) * c_s[r][cc] + sigmoidf_fast(gi) * tanhf(gg);
      const float hn = sigmoidf_fast(go) * tanhf(cn);
      c_s[r][cc] = cn;
      const bf16 hb = (bf16)hn;
      hnxt[r * HDIM + n0 + cc] = hb;              // next step's A operand
      hs[xrow * HDIM + n0 + cc] = hb;             // sequence output for phase 3
    }

    // grid barrier (monotonic counter; double-buffered h makes 1/step safe)
    __syncthreads();
    if (tid == 0) {
      __hip_atomic_fetch_add(ctr, 1u, __ATOMIC_RELEASE, __HIP_MEMORY_SCOPE_AGENT);
      const unsigned target = (unsigned)NWG * (unsigned)(t + 1);
      while (__hip_atomic_load(ctr, __ATOMIC_ACQUIRE, __HIP_MEMORY_SCOPE_AGENT) < target)
        __builtin_amdgcn_s_sleep(1);
    }
    __syncthreads();
    __builtin_amdgcn_fence(__ATOMIC_ACQUIRE, "agent");
  }
}

// ---------------- phase 3: y = hs @ Wo + bo ----------------
// grid: (BS/16, 1), block 256; each wave does a 16x32 strip (8 waves x 32 = 256 cols)
__global__ __launch_bounds__(256) void gemm_out_kernel(const bf16* __restrict__ hsb,
                                                       const bf16* __restrict__ woT,
                                                       const float* __restrict__ bo,
                                                       float* __restrict__ y) {
  const int lane = threadIdx.x & 31, wave = threadIdx.x >> 5;
  const int m0 = blockIdx.x * 16;
  const int n0 = wave * 32;
  const int arow = m0 + (lane & 15);
  const int brow = n0 + (lane & 15);
  v8f acc0 = {}, acc1 = {};
  for (int kb = 0; kb < HDIM; kb += 32) {
    v16bf a  = load_frag(hsb, arow,      HDIM, kb, lane);
    v16bf b0 = load_frag(woT, brow,      HDIM, kb, lane);
    v16bf b1 = load_frag(woT, brow + 16, HDIM, kb, lane);
    acc0 = __builtin_amdgcn_wmma_f32_16x16x32_bf16(false, a, false, b0, (short)0, acc0,
                                                   false, false);
    acc1 = __builtin_amdgcn_wmma_f32_16x16x32_bf16(false, a, false, b1, (short)0, acc1,
                                                   false, false);
  }
  const float bias0 = bo[n0 + (lane & 15)];
  const float bias1 = bo[n0 + 16 + (lane & 15)];
  float* out = y + (size_t)(m0 + ((lane >> 4) << 3)) * ODIM + n0 + (lane & 15);
#pragma unroll
  for (int v = 0; v < 8; ++v) {
    out[(size_t)v * ODIM]      = acc0[v] + bias0;
    out[(size_t)v * ODIM + 16] = acc1[v] + bias1;
  }
}

// ---------------- host ----------------
extern "C" void kernel_launch(void* const* d_in, const int* in_sizes, int n_in,
                              void* d_out, int out_size, void* d_ws, size_t ws_size,
                              hipStream_t stream) {
  const float* x  = (const float*)d_in[0];
  const float* Wx = (const float*)d_in[1];
  const float* Wh = (const float*)d_in[2];
  const float* b  = (const float*)d_in[3];
  const float* Wo = (const float*)d_in[4];
  const float* bo = (const float*)d_in[5];
  float* y = (float*)d_out;

  char* ws = (char*)d_ws;
  size_t off = 0;
  auto alloc = [&](size_t bytes) -> void* {
    void* p = ws + off;
    off = (off + bytes + 255) & ~(size_t)255;
    return p;
  };

  unsigned* ctr = (unsigned*)alloc(256);
  float* xg  = (float*)alloc((size_t)BS * G4H * sizeof(float));     // 512 MiB
  bf16* xb   = (bf16*) alloc((size_t)BS * DIM * sizeof(bf16));      //  32 MiB
  bf16* hsb  = (bf16*) alloc((size_t)BS * HDIM * sizeof(bf16));     //  64 MiB
  bf16* wxT  = (bf16*) alloc((size_t)G4H * DIM * sizeof(bf16));
  bf16* whT  = (bf16*) alloc((size_t)G4H * HDIM * sizeof(bf16));
  bf16* woT  = (bf16*) alloc((size_t)ODIM * HDIM * sizeof(bf16));
  bf16* hbuf = (bf16*) alloc((size_t)2 * BSZ * HDIM * sizeof(bf16));

  init_kernel<<<128, 256, 0, stream>>>(hbuf, ctr);

  const size_t nx = (size_t)BS * DIM;
  cast_bf16_kernel<<<(unsigned)((nx + 255) / 256), 256, 0, stream>>>(x, xb, nx);

  transpose_cast_kernel<<<dim3(G4H / 16, DIM / 16), 256, 0, stream>>>(Wx, wxT, DIM, G4H);
  transpose_cast_kernel<<<dim3(G4H / 16, HDIM / 16), 256, 0, stream>>>(Wh, whT, HDIM, G4H);
  transpose_cast_kernel<<<dim3(ODIM / 16, HDIM / 16), 256, 0, stream>>>(Wo, woT, HDIM, ODIM);

  gemm_xg_kernel<<<dim3(BS / 16, G4H / (16 * 8 * 2)), 256, 0, stream>>>(xb, wxT, xg);

  lstm_kernel<<<NWG, 256, (unsigned)WH_LDS_BYTES, stream>>>(xg, whT, b, hbuf, hsb, ctr);

  gemm_out_kernel<<<dim3(BS / 16, 1), 256, 0, stream>>>(hsb, woT, bo, y);
}
